// GCN_80161269612933
// MI455X (gfx1250) — matrix operand order repaired and checked by reading, not compiled, not run on hardware
//
#include <hip/hip_runtime.h>
#include <hip/hip_bf16.h>

// ---------- types for WMMA ----------
typedef __bf16 v16bf __attribute__((ext_vector_type(16)));
typedef __bf16 v8bf  __attribute__((ext_vector_type(8)));
typedef float  v8f   __attribute__((ext_vector_type(8)));

#define CIN  256  // fixed by the reference model
#define COUT 64   // fixed by the reference model

// float -> bf16 RNE (bit manipulation)
static __device__ __forceinline__ unsigned short f2bf_bits(float f) {
    union { float f; unsigned u; } v; v.f = f;
    unsigned r = v.u + 0x7FFFu + ((v.u >> 16) & 1u);
    return (unsigned short)(r >> 16);
}
static __device__ __forceinline__ float bfbits2f(unsigned short s) {
    union { unsigned u; float f; } o; o.u = ((unsigned)s) << 16;
    return o.f;
}

// ---------- one-time split: x -> (hi, lo) bf16 pair, row-major ----------
__global__ __launch_bounds__(256) void split_bf16_kernel(const float* __restrict__ x,
                                                         unsigned short* __restrict__ hi,
                                                         unsigned short* __restrict__ lo,
                                                         long long n) {
    long long i = (long long)blockIdx.x * 256 + threadIdx.x;
    if (i >= n) return;
    float v = x[i];
    unsigned short h = f2bf_bits(v);
    hi[i] = h;
    lo[i] = f2bf_bits(v - bfbits2f(h));
}

// ---------- utility: zero buffers ----------
__global__ __launch_bounds__(256) void zero_f32v4(float4* __restrict__ p, long long n4) {
    long long i = (long long)blockIdx.x * 256 + threadIdx.x;
    if (i < n4) p[i] = make_float4(0.f, 0.f, 0.f, 0.f);
}
__global__ __launch_bounds__(256) void zero_f32(float* __restrict__ p, long long n) {
    long long i = (long long)blockIdx.x * 256 + threadIdx.x;
    if (i < n) p[i] = 0.0f;
}

// ---------- degree: deg[dst] += 1 per edge ----------
__global__ __launch_bounds__(256) void degree_kernel(const int* __restrict__ ei,
                                                     float* __restrict__ deg, int E) {
    long long i = (long long)blockIdx.x * 256 + threadIdx.x;
    if (i < E) atomicAdd(&deg[ei[(long long)E + i]], 1.0f);
}

// ---------- GEMM: H[N,64] = X[N,256] @ W[64,256]^T, bf16-split WMMA ----------
// Block = 128 threads = 4 waves per 16-row node tile; wave w owns output
// channels [16w, 16w+16). Fragments are straight b128 loads from the
// pre-split row-major bf16 arrays (no conversion VALU in the hot loop).
// Accuracy: Ahi*Bhi + Ahi*Blo + Alo*Bhi with f32 accumulate ~ fp32 GEMM.
__global__ __launch_bounds__(128) void gcn_gemm_wmma(const unsigned short* __restrict__ Xhi,
                                                     const unsigned short* __restrict__ Xlo,
                                                     const unsigned short* __restrict__ Whi,
                                                     const unsigned short* __restrict__ Wlo,
                                                     float* __restrict__ H,
                                                     int N) {
    const int lane = threadIdx.x & 31;
    const int wave = threadIdx.x >> 5;     // output-channel tile
    const int half = lane >> 4;            // lane group 0/1
    const int mn   = lane & 15;

    const int rowBase = blockIdx.x * 16;
    int arow = rowBase + mn; if (arow >= N) arow = N - 1;  // clamp (no EXEC divergence)

    const unsigned short* __restrict__ xh = Xhi + (long long)arow * CIN;
    const unsigned short* __restrict__ xl = Xlo + (long long)arow * CIN;
    const int bcol = wave * 16 + mn;
    const unsigned short* __restrict__ wh = Whi + (long long)bcol * CIN;
    const unsigned short* __restrict__ wl = Wlo + (long long)bcol * CIN;

    const int kA0 = half * 8;    // A: two 8-elem runs at +0 and +16
    const int kB0 = half * 8;    // B: two 8-elem runs at +0 and +16

    v8f acc = {};
#pragma unroll 2
    for (int kb = 0; kb < CIN; kb += 32) {
        v8bf ah0 = *(const v8bf*)(xh + kb + kA0);
        v8bf ah1 = *(const v8bf*)(xh + kb + kA0 + 16);
        v8bf al0 = *(const v8bf*)(xl + kb + kA0);
        v8bf al1 = *(const v8bf*)(xl + kb + kA0 + 16);
        v8bf bh0 = *(const v8bf*)(wh + kb + kB0);
        v8bf bh1 = *(const v8bf*)(wh + kb + kB0 + 16);
        v8bf bl0 = *(const v8bf*)(wl + kb + kB0);
        v8bf bl1 = *(const v8bf*)(wl + kb + kB0 + 16);

        v16bf ahi = __builtin_shufflevector(ah0, ah1, 0,1,2,3,4,5,6,7,8,9,10,11,12,13,14,15);
        v16bf alo = __builtin_shufflevector(al0, al1, 0,1,2,3,4,5,6,7,8,9,10,11,12,13,14,15);
        v16bf bhi = __builtin_shufflevector(bh0, bh1, 0,1,2,3,4,5,6,7,8,9,10,11,12,13,14,15);
        v16bf blo = __builtin_shufflevector(bl0, bl1, 0,1,2,3,4,5,6,7,8,9,10,11,12,13,14,15);

        acc = __builtin_amdgcn_wmma_f32_16x16x32_bf16(false, ahi, false, bhi,
                                                      (short)0, acc, false, false);
        acc = __builtin_amdgcn_wmma_f32_16x16x32_bf16(false, ahi, false, blo,
                                                      (short)0, acc, false, false);
        acc = __builtin_amdgcn_wmma_f32_16x16x32_bf16(false, alo, false, bhi,
                                                      (short)0, acc, false, false);
    }

    // D layout: VGPR r -> row (rowBase + r + 8*laneHi), col = wave*16 + (lane&15).
    // Per-lane base pointer once; rows advance by COUT floats -> immediate offsets.
    float* __restrict__ hp =
        H + (long long)(rowBase + half * 8) * COUT + wave * 16 + mn;
    if (rowBase + 16 <= N) {           // wave-uniform fast path: no exec churn
#pragma unroll
        for (int r = 0; r < 8; ++r) hp[r * COUT] = acc[r];
    } else {                           // tail tile (unused for N % 16 == 0)
#pragma unroll
        for (int r = 0; r < 8; ++r)
            if (rowBase + r + half * 8 < N) hp[r * COUT] = acc[r];
    }
}

// ---------- scatter: agg[dst] += H[src]; 16 threads/edge, float4 gather ----------
__global__ __launch_bounds__(256) void scatter_kernel(const float4* __restrict__ Hv,
                                                      const int* __restrict__ ei,
                                                      float* __restrict__ agg, int E) {
    long long t = (long long)blockIdx.x * 256 + threadIdx.x;
    int e = (int)(t >> 4);
    if (e >= E) return;
    int q = (int)(t & 15);
    int s = ei[e];
    int d = ei[(long long)E + e];
    float4 v = Hv[(long long)s * 16 + q];
    float* ag = agg + (long long)d * COUT + q * 4;
    atomicAdd(ag + 0, v.x);
    atomicAdd(ag + 1, v.y);
    atomicAdd(ag + 2, v.z);
    atomicAdd(ag + 3, v.w);
}

// ---------- finalize: out = relu((agg + H) / (deg + 1)), float4 ----------
__global__ __launch_bounds__(256) void finalize_kernel(const float4* __restrict__ Hv,
                                                       const float4* __restrict__ aggv,
                                                       const float* __restrict__ deg,
                                                       float4* __restrict__ out,
                                                       long long n4) {
    long long i = (long long)blockIdx.x * 256 + threadIdx.x;
    if (i >= n4) return;
    long long node = i >> 4;  // 16 float4 per node (COUT==64)
    float inv = 1.0f / (deg[node] + 1.0f);
    float4 h = Hv[i], a = aggv[i];
    float4 r;
    r.x = fmaxf((a.x + h.x) * inv, 0.0f);
    r.y = fmaxf((a.y + h.y) * inv, 0.0f);
    r.z = fmaxf((a.z + h.z) * inv, 0.0f);
    r.w = fmaxf((a.w + h.w) * inv, 0.0f);
    out[i] = r;
}

extern "C" void kernel_launch(void* const* d_in, const int* in_sizes, int n_in,
                              void* d_out, int out_size, void* d_ws, size_t ws_size,
                              hipStream_t stream) {
    const float* X  = (const float*)d_in[0];
    const float* W  = (const float*)d_in[1];
    const int*   ei = (const int*)d_in[2];

    const int N = in_sizes[0] / CIN;   // 100000
    const int E = in_sizes[2] / 2;     // 1600000

    const long long nx = (long long)N * CIN;
    const long long nw = (long long)COUT * CIN;
    const long long nh = (long long)N * COUT;

    // workspace layout (all chunks 256B-aligned):
    // deg[N] | H[nh] | agg[nh] | Xhi[nx] | Xlo[nx] | Whi[nw] | Wlo[nw]
    char* ws = (char*)d_ws;
    size_t off = 0;
    auto carve = [&](size_t bytes) { char* p = ws + off; off += (bytes + 255) & ~(size_t)255; return p; };
    float*          deg = (float*)carve((size_t)N * 4);
    float*          H   = (float*)carve((size_t)nh * 4);
    float*          agg = (float*)carve((size_t)nh * 4);
    unsigned short* Xhi = (unsigned short*)carve((size_t)nx * 2);
    unsigned short* Xlo = (unsigned short*)carve((size_t)nx * 2);
    unsigned short* Whi = (unsigned short*)carve((size_t)nw * 2);
    unsigned short* Wlo = (unsigned short*)carve((size_t)nw * 2);

    float* out = (float*)d_out;

    // 1) zero deg + agg
    zero_f32<<<(unsigned)((N + 255) / 256), 256, 0, stream>>>(deg, N);
    zero_f32v4<<<(unsigned)((nh / 4 + 255) / 256), 256, 0, stream>>>((float4*)agg, nh / 4);

    // 2) one-time fp32 -> (bf16 hi, bf16 lo) split of X and W
    split_bf16_kernel<<<(unsigned)((nx + 255) / 256), 256, 0, stream>>>(X, Xhi, Xlo, nx);
    split_bf16_kernel<<<(unsigned)((nw + 255) / 256), 256, 0, stream>>>(W, Whi, Wlo, nw);

    // 3) degree
    degree_kernel<<<(unsigned)((E + 255) / 256), 256, 0, stream>>>(ei, deg, E);

    // 4) WMMA GEMM: one block per 16-row tile, 4 waves cover COUT=64
    gcn_gemm_wmma<<<(unsigned)((N + 15) / 16), 128, 0, stream>>>(Xhi, Xlo, Whi, Wlo, H, N);

    // 5) edge scatter-add (dominant phase: ~102M f32 atomics into L2)
    scatter_kernel<<<(unsigned)(((long long)E * 16 + 255) / 256), 256, 0, stream>>>(
        (const float4*)H, ei, agg, E);

    // 6) normalize + relu
    finalize_kernel<<<(unsigned)((nh / 4 + 255) / 256), 256, 0, stream>>>(
        (const float4*)H, (const float4*)agg, deg, (float4*)out, nh / 4);
}
